// KNRM_26087631355983
// MI455X (gfx1250) — matrix operand order repaired and checked by reading, not compiled, not run on hardware
//
#include <hip/hip_runtime.h>
#include <hip/hip_bf16.h>
#include <math.h>

typedef __attribute__((ext_vector_type(16))) _Float16 v16h;
typedef __attribute__((ext_vector_type(8)))  float    v8f;
typedef __attribute__((ext_vector_type(2)))  float    f32x2;

#define NK   21
#define DD   128
#define BS   64
#define LQW  15
#define LQE  10
#define LPW  4096
#define LPE  512
#define PITCH 132      // f16 elems per LDS row (128 + 4 pad -> conflict-free strided gathers)
#define NWAVE 8

// 1/max(sqrt(ss), 1e-8) via hardware v_rsq_f32 (~1 ulp), branchless.
// sqrt(ss) < 1e-8  <=>  ss < 1e-16  -> reference clamps to 1e8.
__device__ __forceinline__ float inv_norm_fast(float ss) {
  float r = __builtin_amdgcn_rsqf(ss);
  return ss > 1e-16f ? r : 1e8f;
}

// ---------------------------------------------------------------------------
// Kernel 1: one workgroup per (batch, psg_type).
//   phase 0: normalize query vectors (f32), store f16 to LDS (zero-padded to 16 rows)
//   phase 1: per wave, stream 16-row passage tiles: raw f16 A-tile in LDS,
//            WMMA f32_16x16x32_f16 against resident B fragments (wrd & ent),
//            post-scale by exact f32 1/||p||, packed-f32 factored soft-binning.
//   phase 2: xor-16 lane reduce -> LDS ds_add_f32 -> masked sum_q log -> phi_raw
// ---------------------------------------------------------------------------
__global__ void __launch_bounds__(256) knrm_phi_kernel(
    const float* __restrict__ qw, const float* __restrict__ pw,
    const float* __restrict__ qe, const float* __restrict__ pe,
    float* __restrict__ phi_raw)
{
  __shared__ _Float16 qbuf[2][16 * PITCH];          // normalized queries (f16)
  __shared__ _Float16 pstage[NWAVE][16 * PITCH];    // raw passage tile per wave (f16)
  __shared__ float    Ksum[2][16][NK];              // cross-wave bin sums

  const int b     = blockIdx.x;
  const int ptype = blockIdx.y;                     // 0: psg_wrd, 1: psg_ent
  const int tid   = threadIdx.x;
  const int wave  = tid >> 5;
  const int lane  = tid & 31;
  const int qn_   = lane & 15;                      // this lane's q column (N)
  const int g     = lane >> 4;                      // half-wave group

  const float BIN = 1.8f / 21.f;
  const float MU0 = 0.9f - BIN * 0.5f;

  // ---- zero cross-wave accumulators ----
  for (int i = tid; i < 2 * 16 * NK; i += 256) (&Ksum[0][0][0])[i] = 0.f;

  // ---- phase 0: queries -> normalized f16 in LDS (rows >= Lq zeroed) ----
  #pragma unroll
  for (int j = 0; j < 4; ++j) {
    int rr = wave + 8 * j;                 // 0..31 logical rows
    int mm = rr >> 4;                      // 0: wrd, 1: ent
    int q  = rr & 15;
    int Lq = mm ? LQE : LQW;
    const float* src = mm ? qe : qw;
    float4 v = make_float4(0.f, 0.f, 0.f, 0.f);
    if (q < Lq) v = ((const float4*)(src + ((size_t)b * Lq + q) * DD))[lane];
    float ss = v.x * v.x + v.y * v.y + v.z * v.z + v.w * v.w;
    #pragma unroll
    for (int off = 16; off >= 1; off >>= 1) ss += __shfl_xor(ss, off, 32);
    float inv = inv_norm_fast(ss);                  // padded rows: 0 * 1e8 = 0
    _Float16* drow = &qbuf[mm][q * PITCH + lane * 4];
    drow[0] = (_Float16)(v.x * inv);
    drow[1] = (_Float16)(v.y * inv);
    drow[2] = (_Float16)(v.z * inv);
    drow[3] = (_Float16)(v.w * inv);
  }
  __syncthreads();

  // ---- resident B fragments (32x16 f16, K chunks of 32) for wrd and ent ----
  v16h bw[4], bent[4];
  #pragma unroll
  for (int mm = 0; mm < 2; ++mm) {
    #pragma unroll
    for (int c = 0; c < 4; ++c) {
      union { v16h h; unsigned u[8]; } f;
      #pragma unroll
      for (int v = 0; v < 8; ++v) {
        int koff = c * 32 + g * 16 + 2 * v;         // lanes0-15: K 0..15, lanes16-31: K 16..31
        f.u[v] = *(const unsigned*)&qbuf[mm][qn_ * PITCH + koff];
      }
      if (mm == 0) bw[c] = f.h; else bent[c] = f.h;
    }
  }

  // ---- factored Gaussian constants: f_k(c) = E(c) * R(c)^k * Ck ----
  float Ck[NK];
  #pragma unroll
  for (int k = 0; k < NK; ++k) {
    float mu = MU0 - BIN * (float)k;
    Ck[k] = __expf(-2.f * (mu * mu - MU0 * MU0));   // compile-time foldable
  }

  // packed accumulators: .x = qry_wrd bins, .y = qry_ent bins
  f32x2 acc2[NK];
  #pragma unroll
  for (int k = 0; k < NK; ++k) acc2[k] = (f32x2){0.f, 0.f};

  const int    Lp    = ptype ? LPE : LPW;
  const float* psrc  = ptype ? pe : pw;
  const int    nTile = Lp >> 4;
  const int    rowA  = lane >> 1;
  const int    halfc = (lane & 1) * 64;
  _Float16*    stage = &pstage[wave][0];

  for (int t = wave; t < nTile; t += NWAVE) {
    const int p0 = t << 4;

    // prefetch this wave's next tile (p0 + 128 rows ahead) into the caches
    if (t + NWAVE < nTile) {
      const char* np = (const char*)(psrc + ((size_t)b * Lp + p0 + 16 * NWAVE + rowA) * DD + halfc);
      __builtin_prefetch(np, 0, 1);
    }

    // coalesced raw-tile load (each lane: contiguous 64 floats), sumsq on the fly
    const float4* rp = (const float4*)(psrc + ((size_t)b * Lp + p0 + rowA) * DD + halfc);
    float ss = 0.f;
    _Float16* drow = &stage[rowA * PITCH + halfc];
    #pragma unroll
    for (int i = 0; i < 16; ++i) {
      float4 v = rp[i];
      ss += v.x * v.x + v.y * v.y + v.z * v.z + v.w * v.w;
      drow[i * 4 + 0] = (_Float16)v.x;
      drow[i * 4 + 1] = (_Float16)v.y;
      drow[i * 4 + 2] = (_Float16)v.z;
      drow[i * 4 + 3] = (_Float16)v.w;
    }
    ss += __shfl_xor(ss, 1, 32);                    // partner holds other half of row
    float inv = inv_norm_fast(ss);                  // v_rsq_f32, exact f32 row scale
    float invA[8];
    #pragma unroll
    for (int r = 0; r < 8; ++r) invA[r] = __shfl(inv, (g * 8 + r) * 2, 32);

    asm volatile("s_wait_dscnt 0" ::: "memory");    // tile staged before gathers

    // K-loop: gather A fragment from LDS, two WMMAs per chunk
    v8f dw = {}, de = {};
    #pragma unroll
    for (int c = 0; c < 4; ++c) {
      union { v16h h; unsigned u[8]; } a;
      #pragma unroll
      for (int v = 0; v < 8; ++v) {
        int koff = c * 32 + g * 16 + v * 2;  // contiguous K: lanes0-15 K 0..15, lanes16-31 K 16..31
        a.u[v] = *(const unsigned*)&stage[qn_ * PITCH + koff];
      }
      dw = __builtin_amdgcn_wmma_f32_16x16x32_f16(false, a.h, false, bw[c],
                                                  (short)0, dw, false, false);
      de = __builtin_amdgcn_wmma_f32_16x16x32_f16(false, a.h, false, bent[c],
                                                  (short)0, de, false, false);
    }

    // packed soft-binning: both matrices in one <2 x float> lane
    // f_k(c) = E * R^k * Ck   with E = exp(-2(c-mu0)^2), R = exp(-4*bin*c)
    #pragma unroll
    for (int r = 0; r < 8; ++r) {
      f32x2 cv; cv.x = dw[r] * invA[r]; cv.y = de[r] * invA[r];
      f32x2 tmu = cv - MU0;
      f32x2 ea  = -2.f * tmu * tmu;
      f32x2 ra  = (-4.f * BIN) * cv;
      f32x2 E, R;
      E.x = __expf(ea.x); E.y = __expf(ea.y);
      R.x = __expf(ra.x); R.y = __expf(ra.y);
      f32x2 wv = E;
      #pragma unroll
      for (int k = 0; k < NK; ++k) {
        acc2[k] += wv * Ck[k];                      // v_pk_fma_f32
        wv *= R;                                    // v_pk_mul_f32
      }
    }
  }

  // ---- lane-pair reduce then cross-wave LDS float atomics ----
  #pragma unroll
  for (int k = 0; k < NK; ++k) {
    float tw = acc2[k].x + __shfl_xor(acc2[k].x, 16, 32);
    float te = acc2[k].y + __shfl_xor(acc2[k].y, 16, 32);
    if (lane < 16) {
      atomicAdd(&Ksum[0][lane][k], tw);
      atomicAdd(&Ksum[1][lane][k], te);
    }
  }
  __syncthreads();

  // ---- masked sum_q log -> phi_raw[b][(2*qtype+ptype)*21 + k] ----
  if (tid < 2 * NK) {
    int qt = tid / NK, k = tid % NK;
    int Lq = qt ? LQE : LQW;
    float s = 0.f;
    for (int q = 0; q < Lq; ++q) s += __logf(Ksum[qt][q][k]);
    phi_raw[(size_t)b * (4 * NK) + (2 * qt + ptype) * NK + k] = s;
  }
}

// ---------------------------------------------------------------------------
// Kernel 2: BatchNorm1d (batch stats, biased var) + linear + tanh. Tiny.
// ---------------------------------------------------------------------------
__global__ void __launch_bounds__(128) knrm_bn_tanh_kernel(
    const float* __restrict__ phi, const float* __restrict__ W,
    const float* __restrict__ bias, const float* __restrict__ gamma,
    const float* __restrict__ beta, float* __restrict__ out)
{
  __shared__ float zmat[BS][4 * NK + 1];
  const int tid = threadIdx.x;
  if (tid < 4 * NK) {
    float s = 0.f, s2 = 0.f;
    for (int bb = 0; bb < BS; ++bb) {
      float x = phi[bb * 4 * NK + tid];
      s += x; s2 += x * x;
    }
    float mean = s * (1.f / BS);
    float var  = s2 * (1.f / BS) - mean * mean;      // biased, matches jnp.var
    float rstd = rsqrtf(var + 1e-5f);
    float gm = gamma[tid], bt = beta[tid], wf = W[tid];
    for (int bb = 0; bb < BS; ++bb) {
      float z = (phi[bb * 4 * NK + tid] - mean) * rstd * gm + bt;
      zmat[bb][tid] = z * wf;
    }
  }
  __syncthreads();
  if (tid < BS) {
    float s = bias[0];
    for (int f = 0; f < 4 * NK; ++f) s += zmat[tid][f];
    out[tid] = tanhf(s);
  }
}

extern "C" void kernel_launch(void* const* d_in, const int* in_sizes, int n_in,
                              void* d_out, int out_size, void* d_ws, size_t ws_size,
                              hipStream_t stream) {
  (void)in_sizes; (void)n_in; (void)out_size; (void)ws_size;
  const float* qw = (const float*)d_in[0];
  const float* pw = (const float*)d_in[1];
  const float* qe = (const float*)d_in[2];
  const float* pe = (const float*)d_in[3];
  const float* W  = (const float*)d_in[4];
  const float* bb = (const float*)d_in[5];
  const float* gm = (const float*)d_in[6];
  const float* bt = (const float*)d_in[7];
  float* phi_raw = (float*)d_ws;                     // 64*84 floats of scratch

  knrm_phi_kernel<<<dim3(BS, 2), 256, 0, stream>>>(qw, pw, qe, pe, phi_raw);
  knrm_bn_tanh_kernel<<<1, 128, 0, stream>>>(phi_raw, W, bb, gm, bt, (float*)d_out);
}
